// LightGlue_36060545417836
// MI455X (gfx1250) — matrix operand order repaired and checked by reading, not compiled, not run on hardware
//
#include <hip/hip_runtime.h>
#include <hip/hip_bf16.h>
#include <math.h>

// ---------------------------------------------------------------------------
// LightGlue forward for MI455X (gfx1250, wave32, WMMA f32_16x16x32_f16).
// Bandwidth-bound -> flash-attention fusion, no materialized attn matrices.
// All LDS operand tiles laid out so every WMMA fragment load is 2x ds_load_b128:
//   A tiles row-major (two contiguous 8-half runs per lane),
//   B tiles stored transposed Bt[n][k] (two contiguous 8-half runs per lane).
// ---------------------------------------------------------------------------

#define Bc   4
#define Mseq 2048
#define Cc   256
#define Hh   4
#define HDd  64
#define Lc   9

typedef __attribute__((ext_vector_type(16))) _Float16 v16h;
typedef __attribute__((ext_vector_type(8)))  _Float16 v8h;
typedef __attribute__((ext_vector_type(2)))  _Float16 v2h;
typedef __attribute__((ext_vector_type(8)))  float    v8f;

__device__ __forceinline__ v16h frag_cat(v8h lo, v8h hi) {
  v16h x;
#pragma unroll
  for (int t = 0; t < 8; ++t) { x[t] = lo[t]; x[8 + t] = hi[t]; }
  return x;
}
// A: 16x32 f16, row-major tile, ld multiple of 8 halves.
// lane(0-15)=row, VGPR pairs: K {half*8..+7} then {16+half*8..+7}.
__device__ __forceinline__ v16h load_frag_a(const _Float16* base, int ld, int lane) {
  const int half = lane >> 4, r = lane & 15;
  const _Float16* p = base + r * ld + half * 8;
  return frag_cat(*(const v8h*)p, *(const v8h*)(p + 16));
}
// B (transposed storage Bt[n][k]): lane(0-15)=col n, K {half*8..+7} then {16+half*8..+7}.
__device__ __forceinline__ v16h load_frag_bt(const _Float16* base, int ld, int lane) {
  const int half = lane >> 4, r = lane & 15;
  const _Float16* p = base + r * ld + half * 8;
  return frag_cat(*(const v8h*)p, *(const v8h*)(p + 16));
}
// C/D: v[i] <-> row (i + 8*half), col (lane&15) within the 16x16 tile.

__device__ __forceinline__ v8f wmma_f16(v16h a, v16h b, v8f c) {
  return __builtin_amdgcn_wmma_f32_16x16x32_f16(false, a, false, b, (short)0, c,
                                                false, false);
}
__device__ __forceinline__ v8h cvt8(float4 a, float4 b) {
  v8h h;
  h[0] = (_Float16)a.x; h[1] = (_Float16)a.y; h[2] = (_Float16)a.z; h[3] = (_Float16)a.w;
  h[4] = (_Float16)b.x; h[5] = (_Float16)b.y; h[6] = (_Float16)b.z; h[7] = (_Float16)b.w;
  return h;
}
__device__ __forceinline__ float logsig(float x) {
  return x >= 0.f ? -log1pf(__expf(-x)) : x - log1pf(__expf(x));
}
__device__ __forceinline__ float gelu_tanh(float x) {
  const float x3 = x * x * x;
  return 0.5f * x * (1.f + tanhf(0.79788456080286535588f * (x + 0.044715f * x3)));
}

// ---------------------------------------------------------------------------
// Generic GEMM: C[r,c] (op)= alpha * (A[MxK] @ W[KxN] + bias[c])
// Block tile 64x128, K step 32 (all dims exact multiples in this model).
// ---------------------------------------------------------------------------
__global__ __launch_bounds__(256) void gemm_wmma_kernel(
    const float* __restrict__ A, int lda, const float* __restrict__ W, int ldw,
    const float* __restrict__ bias, float* __restrict__ C, int ldc, int Kdim,
    float alpha, int accumulate) {
  __shared__ _Float16 As[64][40];
  __shared__ _Float16 Bt[128][40];  // Bt[n][k]
  const int tid = threadIdx.x, lane = tid & 31, w = tid >> 5;
  const int row0 = blockIdx.y * 64, col0 = blockIdx.x * 128;
  const int rb = (w & 3) * 16, cbw = (w >> 2) * 64;
  v8f acc[4];
#pragma unroll
  for (int t = 0; t < 4; ++t) acc[t] = (v8f){0, 0, 0, 0, 0, 0, 0, 0};

  const int ar = tid >> 2, ak = (tid & 3) * 8;          // A staging coords
  const int bn = tid & 127, bk = (tid >> 7) * 16;       // B staging coords

  for (int k0 = 0; k0 < Kdim; k0 += 32) {
    if (k0 + 32 < Kdim)
      __builtin_prefetch(&A[(size_t)(row0 + ar) * lda + k0 + 32], 0, 1);
    // A tile: vector loads + vector LDS stores
    {
      const float* gp = &A[(size_t)(row0 + ar) * lda + k0 + ak];
      const float4 f0 = ((const float4*)gp)[0], f1 = ((const float4*)gp)[1];
      *(v8h*)&As[ar][ak] = cvt8(f0, f1);
    }
    // B tile transposed: coalesced global reads, scalar LDS scatter
#pragma unroll
    for (int kk = 0; kk < 16; ++kk)
      Bt[bn][bk + kk] = (_Float16)W[(size_t)(k0 + bk + kk) * ldw + col0 + bn];
    __syncthreads();
    const v16h a = load_frag_a(&As[rb][0], 40, lane);
#pragma unroll
    for (int t = 0; t < 4; ++t) {
      const v16h b = load_frag_bt(&Bt[cbw + t * 16][0], 40, lane);
      acc[t] = wmma_f16(a, b, acc[t]);
    }
    __syncthreads();
  }
  const int half = lane >> 4, r = lane & 15;
#pragma unroll
  for (int t = 0; t < 4; ++t) {
    const int col = col0 + cbw + t * 16 + r;
    const float bval = bias ? bias[col] : 0.f;
#pragma unroll
    for (int i = 0; i < 8; ++i) {
      const int row = row0 + rb + i + half * 8;
      const float v = alpha * (acc[t][i] + bval);
      const size_t idx = (size_t)row * ldc + col;
      if (accumulate) C[idx] += v; else C[idx] = v;
    }
  }
}

// ---------------------------------------------------------------------------
// Batched NT GEMM: sim[b] = md0[b] @ md1[b]^T into inner 2048x2048 of the
// (2049,2049) scores plane. md1 row-major (n,k) is already Bt layout.
// ---------------------------------------------------------------------------
__global__ __launch_bounds__(256) void gemm_nt_kernel(const float* __restrict__ A,
                                                      const float* __restrict__ Bm,
                                                      float* __restrict__ Cout) {
  __shared__ _Float16 As[64][40];
  __shared__ _Float16 Bt[128][40];
  const int tid = threadIdx.x, lane = tid & 31, w = tid >> 5;
  const int bz = blockIdx.z;
  const int row0 = blockIdx.y * 64, col0 = blockIdx.x * 128;
  const float* Ab = A + (size_t)bz * Mseq * Cc;
  const float* Bb = Bm + (size_t)bz * Mseq * Cc;
  float* Cb = Cout + (size_t)bz * 2049 * 2049;
  const int rb = (w & 3) * 16, cbw = (w >> 2) * 64;
  v8f acc[4];
#pragma unroll
  for (int t = 0; t < 4; ++t) acc[t] = (v8f){0, 0, 0, 0, 0, 0, 0, 0};

  const int ar = tid >> 2, ak = (tid & 3) * 8;
  const int bn = tid >> 1, bk = (tid & 1) * 16;

  for (int k0 = 0; k0 < Cc; k0 += 32) {
    {
      const float* gp = &Ab[(size_t)(row0 + ar) * Cc + k0 + ak];
      const float4 f0 = ((const float4*)gp)[0], f1 = ((const float4*)gp)[1];
      *(v8h*)&As[ar][ak] = cvt8(f0, f1);
    }
    {
      const float* gp = &Bb[(size_t)(col0 + bn) * Cc + k0 + bk];
      const float4 f0 = ((const float4*)gp)[0], f1 = ((const float4*)gp)[1];
      const float4 f2 = ((const float4*)gp)[2], f3 = ((const float4*)gp)[3];
      *(v8h*)&Bt[bn][bk]     = cvt8(f0, f1);
      *(v8h*)&Bt[bn][bk + 8] = cvt8(f2, f3);
    }
    __syncthreads();
    const v16h a = load_frag_a(&As[rb][0], 40, lane);
#pragma unroll
    for (int t = 0; t < 4; ++t) {
      const v16h b = load_frag_bt(&Bt[cbw + t * 16][0], 40, lane);
      acc[t] = wmma_f16(a, b, acc[t]);
    }
    __syncthreads();
  }
  const int half = lane >> 4, r = lane & 15;
#pragma unroll
  for (int t = 0; t < 4; ++t) {
    const int col = col0 + cbw + t * 16 + r;
#pragma unroll
    for (int i = 0; i < 8; ++i)
      Cb[(size_t)(row0 + rb + i + half * 8) * 2049 + col] = acc[t][i];
  }
}

// ---------------------------------------------------------------------------
// Flash attention: one block = 64 query rows for one (b,h). Online softmax,
// O kept in WMMA accumulators across the 2048-key stream.
//   Ks[key][d] row-major == Bt layout for S = Q@K^T  (k-dim = d)
//   Vt[d][key]           == Bt layout for O += P@V   (k-dim = key)
// ---------------------------------------------------------------------------
__global__ __launch_bounds__(256) void flash_attn_kernel(
    const float* __restrict__ Q, int qs, const float* __restrict__ Kp, int ks,
    const float* __restrict__ Vp, int vs, float* __restrict__ O, int os, int nq,
    int nk, float scale) {
  __shared__ _Float16 Qs[64][72];
  __shared__ _Float16 Ks[64][72];
  __shared__ _Float16 Vt[64][72];
  __shared__ _Float16 Ps[64][72];
  __shared__ float    Ss[64][66];
  __shared__ float    fscale[64];
  __shared__ float    linv[64];

  const int tid = threadIdx.x, lane = tid & 31, w = tid >> 5;
  const int half = lane >> 4, r = lane & 15;
  const int bz = blockIdx.y >> 2, h = blockIdx.y & 3;
  const int q0 = blockIdx.x * 64;
  const int rb = (w & 3) * 16, cb = (w >> 2) * 32;
  const int sr = tid >> 2, sd = (tid & 3) * 16;  // row-staging coords

  // stage Q (64x64) as f16, vectorized
  {
    const float* gp = &Q[(size_t)(bz * nq + q0 + sr) * qs + h * HDd + sd];
    const float4 a0 = ((const float4*)gp)[0], a1 = ((const float4*)gp)[1];
    const float4 a2 = ((const float4*)gp)[2], a3 = ((const float4*)gp)[3];
    *(v8h*)&Qs[sr][sd]     = cvt8(a0, a1);
    *(v8h*)&Qs[sr][sd + 8] = cvt8(a2, a3);
  }

  v8f o0 = (v8f){0, 0, 0, 0, 0, 0, 0, 0};
  v8f o1 = (v8f){0, 0, 0, 0, 0, 0, 0, 0};
  float mrow = -3.0e38f, lrow = 0.f;  // valid on tid<64 (row = tid)

  for (int kb = 0; kb < nk; kb += 64) {
    // K tile row-major (vectorized both sides)
    {
      const float* gp = &Kp[(size_t)(bz * nk + kb + sr) * ks + h * HDd + sd];
      const float4 a0 = ((const float4*)gp)[0], a1 = ((const float4*)gp)[1];
      const float4 a2 = ((const float4*)gp)[2], a3 = ((const float4*)gp)[3];
      *(v8h*)&Ks[sr][sd]     = cvt8(a0, a1);
      *(v8h*)&Ks[sr][sd + 8] = cvt8(a2, a3);
    }
    // V transposed: coalesced global reads, scalar LDS scatter
#pragma unroll
    for (int i = 0; i < 16; ++i) {
      const int e = tid + i * 256, n = e >> 6, d = e & 63;
      Vt[d][n] = (_Float16)Vp[(size_t)(bz * nk + kb + n) * vs + h * HDd + d];
    }
    __syncthreads();

    // S tiles (rb, cb) and (rb, cb+16)
    v8f s0 = (v8f){0, 0, 0, 0, 0, 0, 0, 0};
    v8f s1 = (v8f){0, 0, 0, 0, 0, 0, 0, 0};
#pragma unroll
    for (int kc = 0; kc < 64; kc += 32) {
      const v16h a  = load_frag_a(&Qs[rb][kc], 72, lane);
      const v16h b0 = load_frag_bt(&Ks[cb][kc], 72, lane);
      s0 = wmma_f16(a, b0, s0);
      const v16h b1 = load_frag_bt(&Ks[cb + 16][kc], 72, lane);
      s1 = wmma_f16(a, b1, s1);
    }
#pragma unroll
    for (int i = 0; i < 8; ++i) {
      Ss[rb + i + half * 8][cb + r]      = s0[i] * scale;
      Ss[rb + i + half * 8][cb + 16 + r] = s1[i] * scale;
    }
    __syncthreads();

    // online softmax: thread t owns row t
    if (tid < 64) {
      float m_new = mrow;
#pragma unroll 8
      for (int c = 0; c < 64; ++c) m_new = fmaxf(m_new, Ss[tid][c]);
      const float f = __expf(mrow - m_new);
      float s = 0.f;
#pragma unroll 4
      for (int c = 0; c < 64; c += 2) {
        const float p0 = __expf(Ss[tid][c] - m_new);
        const float p1 = __expf(Ss[tid][c + 1] - m_new);
        v2h pp; pp[0] = (_Float16)p0; pp[1] = (_Float16)p1;
        *(v2h*)&Ps[tid][c] = pp;
        s += p0 + p1;
      }
      lrow = lrow * f + s;
      mrow = m_new;
      fscale[tid] = f;
    }
    __syncthreads();

    // rescale accumulators, then O += P @ V
#pragma unroll
    for (int i = 0; i < 8; ++i) {
      const float f = fscale[rb + i + half * 8];
      o0[i] *= f;
      o1[i] *= f;
    }
#pragma unroll
    for (int kc = 0; kc < 64; kc += 32) {
      const v16h a  = load_frag_a(&Ps[rb][kc], 72, lane);
      const v16h b0 = load_frag_bt(&Vt[cb][kc], 72, lane);
      o0 = wmma_f16(a, b0, o0);
      const v16h b1 = load_frag_bt(&Vt[cb + 16][kc], 72, lane);
      o1 = wmma_f16(a, b1, o1);
    }
    __syncthreads();
  }

  if (tid < 64) linv[tid] = 1.f / lrow;
  __syncthreads();
#pragma unroll
  for (int i = 0; i < 8; ++i) {
    const int row = rb + i + half * 8;
    const float il = linv[row];
    const size_t base = (size_t)(bz * nq + q0 + row) * os + h * HDd;
    O[base + cb + r]      = o0[i] * il;
    O[base + cb + 16 + r] = o1[i] * il;
  }
}

// ---------------------------------------------------------------------------
// Elementwise / reduction helpers
// ---------------------------------------------------------------------------
__global__ void copy_kernel(const float* __restrict__ s, float* __restrict__ d,
                            size_t n) {
  const size_t i = (size_t)blockIdx.x * blockDim.x + threadIdx.x;
  if (i < n) d[i] = s[i];
}

__global__ void posenc_kernel(const float* __restrict__ kpts,
                              const float* __restrict__ size,
                              const float* __restrict__ Wr,
                              float* __restrict__ cosb, float* __restrict__ sinb) {
  const int idx = blockIdx.x * blockDim.x + threadIdx.x;  // (b*n+t)*32 + j
  const int j = idx & 31;
  const int bt = idx >> 5;
  const float sc = fmaxf(size[0], size[1]) * 0.5f;
  const float kx = (kpts[bt * 2 + 0] - size[0] * 0.5f) / sc;
  const float ky = (kpts[bt * 2 + 1] - size[1] * 0.5f) / sc;
  const float p = kx * Wr[j] + ky * Wr[32 + j];
  cosb[idx] = cosf(p);
  sinb[idx] = sinf(p);
}

__global__ void rope_kernel(float* __restrict__ qkv,
                            const float* __restrict__ cosb,
                            const float* __restrict__ sinb) {
  const size_t idx = (size_t)blockIdx.x * blockDim.x + threadIdx.x;
  const int d2 = (int)(idx & 31);
  const size_t r1 = idx >> 5;
  const int h = (int)(r1 & 3);
  const size_t bt = r1 >> 2;
  const float c = cosb[bt * 32 + d2], s = sinb[bt * 32 + d2];
  const size_t base = bt * (3 * Cc) + h * HDd + 2 * d2;
  const float q0 = qkv[base], q1 = qkv[base + 1];
  qkv[base] = q0 * c - q1 * s;
  qkv[base + 1] = q1 * c + q0 * s;
  const float k0 = qkv[Cc + base], k1 = qkv[Cc + base + 1];
  qkv[Cc + base] = k0 * c - k1 * s;
  qkv[Cc + base + 1] = k1 * c + k0 * s;
}

__global__ __launch_bounds__(256) void ln_gelu_kernel(float* __restrict__ h,
                                                      const float* __restrict__ g,
                                                      const float* __restrict__ be) {
  const int row = blockIdx.x, tid = threadIdx.x;
  const size_t base = (size_t)row * 512;
  float x0 = h[base + tid], x1 = h[base + 256 + tid];
  float s = x0 + x1, sq = x0 * x0 + x1 * x1;
  __shared__ float rs[8], rq[8];
  for (int o = 16; o; o >>= 1) {
    s += __shfl_xor(s, o, 32);
    sq += __shfl_xor(sq, o, 32);
  }
  if ((tid & 31) == 0) { rs[tid >> 5] = s; rq[tid >> 5] = sq; }
  __syncthreads();
  s = 0.f; sq = 0.f;
#pragma unroll
  for (int i = 0; i < 8; ++i) { s += rs[i]; sq += rq[i]; }
  const float mu = s * (1.f / 512.f);
  const float var = sq * (1.f / 512.f) - mu * mu;
  const float inv = rsqrtf(var + 1e-5f);
  h[base + tid]       = gelu_tanh((x0 - mu) * inv * g[tid] + be[tid]);
  h[base + 256 + tid] = gelu_tanh((x1 - mu) * inv * g[256 + tid] + be[256 + tid]);
}

__global__ __launch_bounds__(256) void zdot_kernel(const float* __restrict__ d,
                                                   const float* __restrict__ wv,
                                                   const float* __restrict__ bb,
                                                   float* __restrict__ z) {
  const int row = blockIdx.x, tid = threadIdx.x;
  float p = d[(size_t)row * Cc + tid] * wv[tid];
  for (int o = 16; o; o >>= 1) p += __shfl_xor(p, o, 32);
  __shared__ float r[8];
  if ((tid & 31) == 0) r[tid >> 5] = p;
  __syncthreads();
  if (tid == 0) {
    float s = 0.f;
#pragma unroll
    for (int i = 0; i < 8; ++i) s += r[i];
    z[row] = s + bb[0];
  }
}

__global__ __launch_bounds__(256) void lse_row_kernel(const float* __restrict__ sc,
                                                      float* __restrict__ lse) {
  const int gid = blockIdx.x, b = gid >> 11, m = gid & 2047, tid = threadIdx.x;
  const float* rp = sc + (size_t)b * 2049 * 2049 + (size_t)m * 2049;
  float mx = -3.0e38f;
  for (int c = tid; c < 2048; c += 256) mx = fmaxf(mx, rp[c]);
  __shared__ float rr[8];
  for (int o = 16; o; o >>= 1) mx = fmaxf(mx, __shfl_xor(mx, o, 32));
  if ((tid & 31) == 0) rr[tid >> 5] = mx;
  __syncthreads();
  float gmx = -3.0e38f;
#pragma unroll
  for (int i = 0; i < 8; ++i) gmx = fmaxf(gmx, rr[i]);
  __syncthreads();
  float s = 0.f;
  for (int c = tid; c < 2048; c += 256) s += __expf(rp[c] - gmx);
  for (int o = 16; o; o >>= 1) s += __shfl_xor(s, o, 32);
  if ((tid & 31) == 0) rr[tid >> 5] = s;
  __syncthreads();
  if (tid == 0) {
    float t = 0.f;
#pragma unroll
    for (int i = 0; i < 8; ++i) t += rr[i];
    lse[gid] = gmx + logf(t);
  }
}

__global__ __launch_bounds__(256) void lse_col_kernel(const float* __restrict__ sc,
                                                      float* __restrict__ lse) {
  const int gid = blockIdx.x, b = gid >> 11, n = gid & 2047, tid = threadIdx.x;
  const float* cp = sc + (size_t)b * 2049 * 2049 + n;
  float mx = -3.0e38f;
  for (int m = tid; m < 2048; m += 256) mx = fmaxf(mx, cp[(size_t)m * 2049]);
  __shared__ float rr[8];
  for (int o = 16; o; o >>= 1) mx = fmaxf(mx, __shfl_xor(mx, o, 32));
  if ((tid & 31) == 0) rr[tid >> 5] = mx;
  __syncthreads();
  float gmx = -3.0e38f;
#pragma unroll
  for (int i = 0; i < 8; ++i) gmx = fmaxf(gmx, rr[i]);
  __syncthreads();
  float s = 0.f;
  for (int m = tid; m < 2048; m += 256) s += __expf(cp[(size_t)m * 2049] - gmx);
  for (int o = 16; o; o >>= 1) s += __shfl_xor(s, o, 32);
  if ((tid & 31) == 0) rr[tid >> 5] = s;
  __syncthreads();
  if (tid == 0) {
    float t = 0.f;
#pragma unroll
    for (int i = 0; i < 8; ++i) t += rr[i];
    lse[gid] = gmx + logf(t);
  }
}

__global__ void score_fin_kernel(float* __restrict__ sc,
                                 const float* __restrict__ lser,
                                 const float* __restrict__ lsec,
                                 const float* __restrict__ z0,
                                 const float* __restrict__ z1) {
  const size_t idx = (size_t)blockIdx.x * blockDim.x + threadIdx.x;
  const int n = (int)(idx & 2047);
  const size_t t2 = idx >> 11;
  const int m = (int)(t2 & 2047);
  const int b = (int)(t2 >> 11);
  const size_t p = (size_t)b * 2049 * 2049 + (size_t)m * 2049 + n;
  sc[p] = sc[p] - lser[b * 2048 + m] - lsec[b * 2048 + n] +
          logsig(z0[b * 2048 + m]) + logsig(z1[b * 2048 + n]);
}

__global__ void edge_kernel(float* __restrict__ sc, const float* __restrict__ z0,
                            const float* __restrict__ z1) {
  const int idx = blockIdx.x * blockDim.x + threadIdx.x;
  if (idx >= Bc * 2049) return;
  const int i = idx % 2049, b = idx / 2049;
  float* sp = sc + (size_t)b * 2049 * 2049;
  if (i < 2048) {
    sp[(size_t)i * 2049 + 2048]    = logsig(-z0[b * 2048 + i]);
    sp[(size_t)2048 * 2049 + i]    = logsig(-z1[b * 2048 + i]);
  } else {
    sp[(size_t)2048 * 2049 + 2048] = 0.f;
  }
}

__global__ __launch_bounds__(256) void rowargmax_kernel(const float* __restrict__ sc,
                                                        float* __restrict__ mx,
                                                        int* __restrict__ ar) {
  const int gid = blockIdx.x, b = gid >> 11, m = gid & 2047, tid = threadIdx.x;
  const float* rp = sc + (size_t)b * 2049 * 2049 + (size_t)m * 2049;
  float bv = -3.0e38f; int bi = 2048;
  for (int c = tid; c < 2048; c += 256) {
    const float v = rp[c];
    if (v > bv || (v == bv && c < bi)) { bv = v; bi = c; }
  }
  __shared__ float sv[256]; __shared__ int si[256];
  sv[tid] = bv; si[tid] = bi;
  __syncthreads();
  for (int o = 128; o; o >>= 1) {
    if (tid < o) {
      const float v2 = sv[tid + o]; const int i2 = si[tid + o];
      if (v2 > sv[tid] || (v2 == sv[tid] && i2 < si[tid])) { sv[tid] = v2; si[tid] = i2; }
    }
    __syncthreads();
  }
  if (tid == 0) { mx[gid] = sv[0]; ar[gid] = si[0]; }
}

__global__ __launch_bounds__(256) void colargmax_kernel(const float* __restrict__ sc,
                                                        int* __restrict__ ar) {
  const int gid = blockIdx.x, b = gid >> 11, n = gid & 2047, tid = threadIdx.x;
  const float* cp = sc + (size_t)b * 2049 * 2049 + n;
  float bv = -3.0e38f; int bi = 2048;
  for (int m = tid; m < 2048; m += 256) {
    const float v = cp[(size_t)m * 2049];
    if (v > bv || (v == bv && m < bi)) { bv = v; bi = m; }
  }
  __shared__ float sv[256]; __shared__ int si[256];
  sv[tid] = bv; si[tid] = bi;
  __syncthreads();
  for (int o = 128; o; o >>= 1) {
    if (tid < o) {
      const float v2 = sv[tid + o]; const int i2 = si[tid + o];
      if (v2 > sv[tid] || (v2 == sv[tid] && i2 < si[tid])) { sv[tid] = v2; si[tid] = i2; }
    }
    __syncthreads();
  }
  if (tid == 0) ar[gid] = si[0];
}

__global__ void match0_kernel(const float* __restrict__ mx, const int* __restrict__ a0,
                              const int* __restrict__ a1, float* __restrict__ m0o,
                              float* __restrict__ ms0o, int* __restrict__ v0o) {
  const int idx = blockIdx.x * blockDim.x + threadIdx.x;
  const int b = idx >> 11, m = idx & 2047;
  const int am = a0[idx];
  const bool mut = (a1[b * 2048 + am] == m);
  const float ms = mut ? __expf(mx[idx]) : 0.f;
  const bool v = mut && (ms > 0.1f);
  m0o[idx] = v ? (float)am : -1.f;
  ms0o[idx] = ms;
  v0o[idx] = v ? 1 : 0;
}

__global__ void match1_kernel(const int* __restrict__ a0, const int* __restrict__ a1,
                              const float* __restrict__ ms0o,
                              const int* __restrict__ v0o, float* __restrict__ m1o,
                              float* __restrict__ ms1o) {
  const int idx = blockIdx.x * blockDim.x + threadIdx.x;
  const int b = idx >> 11, n = idx & 2047;
  const int an = a1[idx];
  const bool mut = (a0[b * 2048 + an] == n);
  const float ms = mut ? ms0o[b * 2048 + an] : 0.f;
  const bool v = mut && (v0o[b * 2048 + an] != 0);
  m1o[idx] = v ? (float)an : -1.f;
  ms1o[idx] = ms;
}

// ---------------------------------------------------------------------------
extern "C" void kernel_launch(void* const* d_in, const int* in_sizes, int n_in,
                              void* d_out, int out_size, void* d_ws, size_t ws_size,
                              hipStream_t stream) {
  (void)in_sizes; (void)n_in; (void)out_size; (void)ws_size;
  const float* kpts0   = (const float*)d_in[0];
  const float* desc0   = (const float*)d_in[1];
  const float* kpts1   = (const float*)d_in[2];
  const float* desc1   = (const float*)d_in[3];
  const float* imsize  = (const float*)d_in[4];
  const float* Wr      = (const float*)d_in[5];
  const float* saqkv_W = (const float*)d_in[6];
  const float* saqkv_b = (const float*)d_in[7];
  const float* saout_W = (const float*)d_in[8];
  const float* saout_b = (const float*)d_in[9];
  const float* saff_W1 = (const float*)d_in[10];
  const float* saff_b1 = (const float*)d_in[11];
  const float* saff_g  = (const float*)d_in[12];
  const float* saff_be = (const float*)d_in[13];
  const float* saff_W2 = (const float*)d_in[14];
  const float* saff_b2 = (const float*)d_in[15];
  const float* caqk_W  = (const float*)d_in[16];
  const float* caqk_b  = (const float*)d_in[17];
  const float* cav_W   = (const float*)d_in[18];
  const float* cav_b   = (const float*)d_in[19];
  const float* caout_W = (const float*)d_in[20];
  const float* caout_b = (const float*)d_in[21];
  const float* caff_W1 = (const float*)d_in[22];
  const float* caff_b1 = (const float*)d_in[23];
  const float* caff_g  = (const float*)d_in[24];
  const float* caff_be = (const float*)d_in[25];
  const float* caff_W2 = (const float*)d_in[26];
  const float* caff_b2 = (const float*)d_in[27];
  const float* fproj_W = (const float*)d_in[28];
  const float* fproj_b = (const float*)d_in[29];
  const float* match_w = (const float*)d_in[30];
  const float* match_b = (const float*)d_in[31];

  const int NR = Bc * Mseq;  // 8192 rows
  float* ws = (float*)d_ws;
  size_t off = 0;
  auto alloc = [&](size_t n) { float* p = ws + off; off += n; return p; };
  float* cos0 = alloc((size_t)NR * 32);
  float* sin0 = alloc((size_t)NR * 32);
  float* cos1 = alloc((size_t)NR * 32);
  float* sin1 = alloc((size_t)NR * 32);
  float* dd0  = alloc((size_t)NR * Cc);
  float* dd1  = alloc((size_t)NR * Cc);
  float* qkv  = alloc((size_t)NR * 3 * Cc);
  float* msg  = alloc((size_t)NR * Cc);
  float* msg2 = alloc((size_t)NR * Cc);
  float* hbuf = alloc((size_t)NR * 2 * Cc);
  float* qk0  = alloc((size_t)NR * Cc);
  float* qk1  = alloc((size_t)NR * Cc);
  float* v0b  = alloc((size_t)NR * Cc);
  float* v1b  = alloc((size_t)NR * Cc);
  float* z0   = alloc(NR);
  float* z1   = alloc(NR);
  float* lser = alloc(NR);
  float* lsec = alloc(NR);
  float* mx0  = alloc(NR);
  int* arg0   = (int*)alloc(NR);
  int* arg1   = (int*)alloc(NR);
  int* valid0 = (int*)alloc(NR);

  float* out_m0  = (float*)d_out;
  float* out_m1  = out_m0 + NR;
  float* out_ms0 = out_m1 + NR;
  float* out_ms1 = out_ms0 + NR;
  float* scores  = out_ms1 + NR;  // (B, 2049, 2049)

  auto gemm = [&](const float* A, int lda, const float* W, int ldw,
                  const float* bias, float* C, int ldc, int Mr, int Kd, int Nd,
                  float alpha, int acc) {
    dim3 g(Nd / 128, Mr / 64);
    gemm_wmma_kernel<<<g, 256, 0, stream>>>(A, lda, W, ldw, bias, C, ldc, Kd,
                                            alpha, acc);
  };
  auto flash = [&](const float* Q, int qs, const float* K, int ks, const float* V,
                   int vs, float* Oo, int os) {
    flash_attn_kernel<<<dim3(Mseq / 64, Bc * Hh), 256, 0, stream>>>(
        Q, qs, K, ks, V, vs, Oo, os, Mseq, Mseq, 0.125f);
  };
  auto ffn = [&](float* x, const float* W1, const float* b1, const float* g,
                 const float* be, const float* W2, const float* b2) {
    gemm(x, Cc, W1, 2 * Cc, b1, hbuf, 2 * Cc, NR, Cc, 2 * Cc, 1.f, 0);
    gemm(msg2, Cc, W1 + (size_t)Cc * 2 * Cc, 2 * Cc, nullptr, hbuf, 2 * Cc, NR,
         Cc, 2 * Cc, 1.f, 1);
    ln_gelu_kernel<<<NR, 256, 0, stream>>>(hbuf, g, be);
    gemm(hbuf, 2 * Cc, W2, Cc, b2, x, Cc, NR, 2 * Cc, Cc, 1.f, 1);
  };
  auto self_block = [&](float* x, const float* cb, const float* sb, int i) {
    gemm(x, Cc, saqkv_W + (size_t)i * Cc * 3 * Cc, 3 * Cc,
         saqkv_b + (size_t)i * 3 * Cc, qkv, 3 * Cc, NR, Cc, 3 * Cc, 1.f, 0);
    rope_kernel<<<(NR * Hh * 32) / 256, 256, 0, stream>>>(qkv, cb, sb);
    flash(qkv, 3 * Cc, qkv + Cc, 3 * Cc, qkv + 2 * Cc, 3 * Cc, msg, Cc);
    gemm(msg, Cc, saout_W + (size_t)i * Cc * Cc, Cc, saout_b + (size_t)i * Cc,
         msg2, Cc, NR, Cc, Cc, 1.f, 0);
    ffn(x, saff_W1 + (size_t)i * 2 * Cc * 2 * Cc, saff_b1 + (size_t)i * 2 * Cc,
        saff_g + (size_t)i * 2 * Cc, saff_be + (size_t)i * 2 * Cc,
        saff_W2 + (size_t)i * 2 * Cc * Cc, saff_b2 + (size_t)i * Cc);
  };

  copy_kernel<<<(NR * Cc) / 256, 256, 0, stream>>>(desc0, dd0, (size_t)NR * Cc);
  copy_kernel<<<(NR * Cc) / 256, 256, 0, stream>>>(desc1, dd1, (size_t)NR * Cc);
  posenc_kernel<<<(NR * 32) / 256, 256, 0, stream>>>(kpts0, imsize, Wr, cos0, sin0);
  posenc_kernel<<<(NR * 32) / 256, 256, 0, stream>>>(kpts1, imsize, Wr, cos1, sin1);

  for (int i = 0; i < Lc; ++i) {
    self_block(dd0, cos0, sin0, i);
    self_block(dd1, cos1, sin1, i);
    // cross block
    const float* Wqk = caqk_W + (size_t)i * Cc * Cc;
    const float* bqk = caqk_b + (size_t)i * Cc;
    const float* Wv  = cav_W + (size_t)i * Cc * Cc;
    const float* bv  = cav_b + (size_t)i * Cc;
    const float* Wo  = caout_W + (size_t)i * Cc * Cc;
    const float* bo  = caout_b + (size_t)i * Cc;
    gemm(dd0, Cc, Wqk, Cc, bqk, qk0, Cc, NR, Cc, Cc, 1.f, 0);
    gemm(dd1, Cc, Wqk, Cc, bqk, qk1, Cc, NR, Cc, Cc, 1.f, 0);
    gemm(dd0, Cc, Wv, Cc, bv, v0b, Cc, NR, Cc, Cc, 1.f, 0);
    gemm(dd1, Cc, Wv, Cc, bv, v1b, Cc, NR, Cc, Cc, 1.f, 0);
    // m0 = softmax_rows(qk0 qk1^T) @ v1   (HD^-0.5 folded into flash scale)
    flash(qk0, Cc, qk1, Cc, v1b, Cc, msg, Cc);
    gemm(msg, Cc, Wo, Cc, bo, msg2, Cc, NR, Cc, Cc, 1.f, 0);
    ffn(dd0, caff_W1 + (size_t)i * 2 * Cc * 2 * Cc, caff_b1 + (size_t)i * 2 * Cc,
        caff_g + (size_t)i * 2 * Cc, caff_be + (size_t)i * 2 * Cc,
        caff_W2 + (size_t)i * 2 * Cc * Cc, caff_b2 + (size_t)i * Cc);
    // m1 = softmax_cols(sim) @ v0 == flash with roles swapped
    flash(qk1, Cc, qk0, Cc, v0b, Cc, msg, Cc);
    gemm(msg, Cc, Wo, Cc, bo, msg2, Cc, NR, Cc, Cc, 1.f, 0);
    ffn(dd1, caff_W1 + (size_t)i * 2 * Cc * 2 * Cc, caff_b1 + (size_t)i * 2 * Cc,
        caff_g + (size_t)i * 2 * Cc, caff_be + (size_t)i * 2 * Cc,
        caff_W2 + (size_t)i * 2 * Cc * Cc, caff_b2 + (size_t)i * Cc);
  }

  // matching head: md = (d @ fproj + b) / C^0.25  (alpha = 0.25)
  gemm(dd0, Cc, fproj_W, Cc, fproj_b, qk0, Cc, NR, Cc, Cc, 0.25f, 0);
  gemm(dd1, Cc, fproj_W, Cc, fproj_b, qk1, Cc, NR, Cc, Cc, 0.25f, 0);
  gemm_nt_kernel<<<dim3(Mseq / 128, Mseq / 64, Bc), 256, 0, stream>>>(qk0, qk1,
                                                                      scores);
  zdot_kernel<<<NR, 256, 0, stream>>>(dd0, match_w, match_b, z0);
  zdot_kernel<<<NR, 256, 0, stream>>>(dd1, match_w, match_b, z1);
  lse_row_kernel<<<NR, 256, 0, stream>>>(scores, lser);
  lse_col_kernel<<<NR, 256, 0, stream>>>(scores, lsec);
  score_fin_kernel<<<(unsigned)(((size_t)Bc * Mseq * Mseq) / 256), 256, 0, stream>>>(
      scores, lser, lsec, z0, z1);
  edge_kernel<<<(Bc * 2049 + 255) / 256, 256, 0, stream>>>(scores, z0, z1);
  rowargmax_kernel<<<NR, 256, 0, stream>>>(scores, mx0, arg0);
  colargmax_kernel<<<NR, 256, 0, stream>>>(scores, arg1);
  match0_kernel<<<NR / 256, 256, 0, stream>>>(mx0, arg0, arg1, out_m0, out_ms0,
                                              valid0);
  match1_kernel<<<NR / 256, 256, 0, stream>>>(arg0, arg1, out_ms0, valid0, out_m1,
                                              out_ms1);
}
